// Decoder_45079976739229
// MI455X (gfx1250) — matrix operand order repaired and checked by reading, not compiled, not run on hardware
//
#include <hip/hip_runtime.h>

// ---------------------------------------------------------------------------
// MoChA decoder for MI455X (gfx1250, wave32, WMMA).
// - All weights converted to bf16 once (~100MB -> resident in 192MB L2, so the
//   48-step scan streams weights at L2 bandwidth, not HBM).
// - Every GEMM runs through v_wmma_f32_16x16x32_bf16, f32 accumulation.
//   B=16 == one WMMA M-tile; each wave computes TWO adjacent 16-col N-tiles
//   sharing one A fragment (halves A traffic, 2 independent WMMAs in flight).
// - T=256 alpha/beta recursion via wave32 shfl prefix scans.
// - WGP-scope global_prefetch_b8 on the weight stream.
// ---------------------------------------------------------------------------

#define T_ 256
#define B_ 16
#define E_ 512
#define H_ 2048
#define N_ 14839
#define U_ 48
#define UB_ (U_ * B_)     // 768
#define EPS_ 1e-6f

typedef __attribute__((ext_vector_type(16))) __bf16 v16bf;
typedef __attribute__((ext_vector_type(8)))  float  v8f;

union Frag { unsigned int u[8]; v16bf v; };

__device__ __forceinline__ unsigned short f2bf(float f) {
    unsigned int u = __float_as_uint(f);
    u += 0x7FFFu + ((u >> 16) & 1u);          // round-to-nearest-even
    return (unsigned short)(u >> 16);
}
__device__ __forceinline__ float sigm(float x) { return 1.0f / (1.0f + expf(-x)); }

__device__ __forceinline__ float scan_incl_add(float v, int lane) {
    #pragma unroll
    for (int d = 1; d < 32; d <<= 1) { float t = __shfl_up(v, d, 32); if (lane >= d) v += t; }
    return v;
}
__device__ __forceinline__ float scan_incl_mul(float v, int lane) {
    #pragma unroll
    for (int d = 1; d < 32; d <<= 1) { float t = __shfl_up(v, d, 32); if (lane >= d) v *= t; }
    return v;
}

// -------------------------- conversion / prep kernels ----------------------

__global__ void k_f32_to_bf16(const float* __restrict__ src, unsigned short* __restrict__ dst, int n) {
    for (int i = blockIdx.x * blockDim.x + threadIdx.x; i < n; i += gridDim.x * blockDim.x)
        dst[i] = f2bf(src[i]);
}

// Wcat[j, 0:1024] = W_ih[j,:]; Wcat[j, 1024:3072] = W_hh[j,:]   (8192 x 3072)
__global__ void k_build_wcat(const float* __restrict__ Wih, const float* __restrict__ Whh,
                             unsigned short* __restrict__ Wcat, int total) {
    for (int i = blockIdx.x * blockDim.x + threadIdx.x; i < total; i += gridDim.x * blockDim.x) {
        int j = i / 3072, k = i % 3072;
        float v = (k < 1024) ? Wih[(size_t)j * 1024 + k] : Whh[(size_t)j * 2048 + (k - 1024)];
        Wcat[i] = f2bf(v);
    }
}

// emb_bf[u*B+b, e] = bf16(emb_table[label[u,b], e])
__global__ void k_gather_emb(const int* __restrict__ label, const float* __restrict__ table,
                             unsigned short* __restrict__ dst, int total) {
    for (int i = blockIdx.x * blockDim.x + threadIdx.x; i < total; i += gridDim.x * blockDim.x) {
        int r = i >> 9, e = i & 511;
        dst[i] = f2bf(table[(size_t)label[r] * E_ + e]);
    }
}

__global__ void k_init(float* __restrict__ h, float* __restrict__ c,
                       float* __restrict__ ctx, float* __restrict__ alpha) {
    int i = blockIdx.x * blockDim.x + threadIdx.x;              // 32768 threads
    if (i < B_ * H_) { h[i] = 0.f; c[i] = 0.f; }
    if (i < B_ * E_) ctx[i] = 0.f;
    if (i < T_ * B_) alpha[i] = (i < B_) ? 1.f : 0.f;           // alpha[t=0,b]=1
}

// zin[b, 0:512]=emb_u ; [512:1024]=ctx(prev) ; [1024:3072]=h(prev)   (bf16)
__global__ void k_build_zin(const unsigned short* __restrict__ embu, const float* __restrict__ ctx,
                            const float* __restrict__ h, unsigned short* __restrict__ zin) {
    int i = blockIdx.x * blockDim.x + threadIdx.x;
    if (i >= B_ * 3072) return;
    int b = i / 3072, k = i % 3072;
    unsigned short v;
    if (k < 512)        v = embu[b * 512 + k];
    else if (k < 1024)  v = f2bf(ctx[b * 512 + (k - 512)]);
    else                v = f2bf(h[b * 2048 + (k - 1024)]);
    zin[i] = v;
}

// ------------------------------ WMMA GEMM ----------------------------------
// C(16 x N per m-tile) = A(16 x K, bf16, row-major, stride lda) x W^T
// W is (N x K) bf16 row-major (stride ldw). Acc f32.
// One wave computes TWO adjacent 16-col tiles sharing one A fragment.
// mode 0: out[(mbase+m)*ldc + n]        (row-major)
// mode 1: out[n*ldc + mbase + m]        (output-transposed, for (N, U*B) layout)
__global__ void k_gemm_bf16(const unsigned short* __restrict__ A, int lda,
                            const unsigned short* __restrict__ W, int ldw,
                            int N, int K, const float* __restrict__ bias,
                            float* __restrict__ out, int mode, int ldc, int mbase0) {
    const int lane   = threadIdx.x & 31;
    const int wv     = threadIdx.x >> 5;
    const int ntile0 = (blockIdx.x * (blockDim.x >> 5) + wv) * 2;
    if (ntile0 * 16 >= N) return;                      // wave-uniform
    const int mtile = blockIdx.y;
    A += (size_t)mtile * 16 * lda;
    const int mbase = mbase0 + mtile * 16;

    const int m    = lane & 15;
    const int kg   = (lane >> 4) * 8;                  // unused; A uses kbh like B
    const int kbh  = (lane >> 4) * 16;                 // B k-half offset
    const int ncol0 = ntile0 * 16 + (lane & 15);
    const int ncol1 = ncol0 + 16;
    const int ncl0  = (ncol0 < N) ? ncol0 : (N - 1);
    const int ncl1  = (ncol1 < N) ? ncol1 : (N - 1);

    int koff[8];
    #pragma unroll
    for (int v = 0; v < 8; ++v)
        koff[v] = kbh + 2 * v;

    const unsigned int*   ap    = reinterpret_cast<const unsigned int*>(A + (size_t)m * lda);
    const unsigned short* wrow0 = W + (size_t)ncl0 * ldw + kbh;
    const unsigned short* wrow1 = W + (size_t)ncl1 * ldw + kbh;

    v8f acc0 = {0.f, 0.f, 0.f, 0.f, 0.f, 0.f, 0.f, 0.f};
    v8f acc1 = {0.f, 0.f, 0.f, 0.f, 0.f, 0.f, 0.f, 0.f};
    for (int kk = 0; kk < K; kk += 32) {
        Frag af, b0, b1;
        #pragma unroll
        for (int v = 0; v < 8; ++v) af.u[v] = ap[(kk + koff[v]) >> 1];
        const unsigned int* wq0 = reinterpret_cast<const unsigned int*>(wrow0 + kk);
        const unsigned int* wq1 = reinterpret_cast<const unsigned int*>(wrow1 + kk);
        #pragma unroll
        for (int v = 0; v < 8; ++v) { b0.u[v] = wq0[v]; b1.u[v] = wq1[v]; }
        if (kk + 32 < K) {                              // WGP-scope prefetch of next slice
            __builtin_prefetch(wrow0 + kk + 64, 0, 3);  // global_prefetch_b8 (near scope)
            __builtin_prefetch(wrow1 + kk + 64, 0, 3);
        }
        acc0 = __builtin_amdgcn_wmma_f32_16x16x32_bf16(false, af.v, false, b0.v,
                                                       (short)0, acc0, false, false);
        acc1 = __builtin_amdgcn_wmma_f32_16x16x32_bf16(false, af.v, false, b1.v,
                                                       (short)0, acc1, false, false);
    }

    const int half = lane >> 4;
    if (ncol0 < N) {
        const float bv = bias ? bias[ncol0] : 0.f;
        #pragma unroll
        for (int r = 0; r < 8; ++r) {
            int mr = r + 8 * half;
            float val = acc0[r] + bv;
            if (mode == 0) out[(size_t)(mbase + mr) * ldc + ncol0] = val;
            else           out[(size_t)ncol0 * ldc + mbase + mr]   = val;
        }
    }
    if (ncol1 < N) {
        const float bv = bias ? bias[ncol1] : 0.f;
        #pragma unroll
        for (int r = 0; r < 8; ++r) {
            int mr = r + 8 * half;
            float val = acc1[r] + bv;
            if (mode == 0) out[(size_t)(mbase + mr) * ldc + ncol1] = val;
            else           out[(size_t)ncol1 * ldc + mbase + mr]   = val;
        }
    }
}

// --------------------------- per-step pointwise ----------------------------

__global__ void k_lstm_pointwise(const float* __restrict__ gates, float* __restrict__ h,
                                 float* __restrict__ c, unsigned short* __restrict__ h_bf) {
    int i = blockIdx.x * blockDim.x + threadIdx.x;     // B*H = 32768
    if (i >= B_ * H_) return;
    int b = i >> 11, j = i & 2047;
    const float* g = gates + (size_t)b * 4 * H_;
    float ig = sigm(g[j]);
    float fg = sigm(g[H_ + j]);
    float gg = tanhf(g[2 * H_ + j]);
    float og = sigm(g[3 * H_ + j]);
    float cn = fg * c[i] + ig * gg;
    float hn = og * tanhf(cn);
    c[i] = cn; h[i] = hn; h_bf[i] = f2bf(hn);
}

// e_raw[t,b] = v_m . tanh(sm[b]+xh_m[t,b]) ; u_raw likewise. One wave per (t,b).
__global__ void k_energies(const float* __restrict__ sm, const float* __restrict__ sc,
                           const float* __restrict__ xhm, const float* __restrict__ xhc,
                           const float* __restrict__ vm, const float* __restrict__ vc,
                           float* __restrict__ e_raw, float* __restrict__ u_raw) {
    int wid  = blockIdx.x * (blockDim.x >> 5) + (threadIdx.x >> 5);
    int lane = threadIdx.x & 31;
    if (wid >= T_ * B_) return;
    int b = wid & 15;
    const float* xm  = xhm + (size_t)wid * E_;
    const float* xc  = xhc + (size_t)wid * E_;
    const float* smb = sm + b * E_;
    const float* scb = sc + b * E_;
    float am = 0.f, ac = 0.f;
    for (int e = lane; e < E_; e += 32) {
        am += vm[e] * tanhf(smb[e] + xm[e]);
        ac += vc[e] * tanhf(scb[e] + xc[e]);
    }
    #pragma unroll
    for (int d = 16; d; d >>= 1) { am += __shfl_xor(am, d, 32); ac += __shfl_xor(ac, d, 32); }
    if (lane == 0) { e_raw[wid] = am; u_raw[wid] = ac; }
}

// MoChA alpha/beta recursion: 1 block, 16 waves (wave = batch), 8 t per lane.
__global__ void k_alpha_beta(const float* __restrict__ e_raw, const float* __restrict__ u_raw,
                             const float* __restrict__ att_mask, const float* __restrict__ r_m,
                             const float* __restrict__ alpha_prev,
                             float* __restrict__ alpha_new, float* __restrict__ beta) {
    const int lane = threadIdx.x & 31;
    const int b    = threadIdx.x >> 5;          // 16 waves
    const int t0   = lane * 8;
    const float rm = r_m[0];

    float p[8], msk[8], ap[8], uu[8];
    #pragma unroll
    for (int i = 0; i < 8; ++i) {
        int t = t0 + i;
        float e = e_raw[t * B_ + b] + rm;
        msk[i] = att_mask[b * T_ + t];
        p[i]   = msk[i] * sigm(e);
        ap[i]  = alpha_prev[t * B_ + b];
        uu[i]  = u_raw[t * B_ + b];
    }
    // exclusive cumprod of (1-p)
    float cp[8]; float run = 1.f;
    #pragma unroll
    for (int i = 0; i < 8; ++i) { cp[i] = run; run *= (1.f - p[i]); }
    float vmul = scan_incl_mul(run, lane);
    float emul = __shfl_up(vmul, 1, 32); if (lane == 0) emul = 1.f;
    #pragma unroll
    for (int i = 0; i < 8; ++i) cp[i] *= emul;
    // alpha = p * cp * cumsum(alpha_prev / max(cp,eps))
    float cum[8]; float rs = 0.f;
    #pragma unroll
    for (int i = 0; i < 8; ++i) { rs += ap[i] / fmaxf(cp[i], EPS_); cum[i] = rs; }
    float vadd = scan_incl_add(rs, lane);
    float eadd = __shfl_up(vadd, 1, 32); if (lane == 0) eadd = 0.f;
    float al[8];
    #pragma unroll
    for (int i = 0; i < 8; ++i) al[i] = p[i] * cp[i] * (cum[i] + eadd);
    // eu = exp(u - max)*mask ; cs = cumsum(eu)
    float mx = uu[0];
    #pragma unroll
    for (int i = 1; i < 8; ++i) mx = fmaxf(mx, uu[i]);
    #pragma unroll
    for (int d = 16; d; d >>= 1) mx = fmaxf(mx, __shfl_xor(mx, d, 32));
    float eu[8], cs[8]; rs = 0.f;
    #pragma unroll
    for (int i = 0; i < 8; ++i) { eu[i] = expf(uu[i] - mx) * msk[i]; rs += eu[i]; cs[i] = rs; }
    vadd = scan_incl_add(rs, lane);
    eadd = __shfl_up(vadd, 1, 32); if (lane == 0) eadd = 0.f;
    #pragma unroll
    for (int i = 0; i < 8; ++i) cs[i] += eadd;
    // denom[t] = cs[t] - cs[t-8]  (CHUNK_W==8 == elems per lane)
    float rk[8], cr[8]; rs = 0.f;
    #pragma unroll
    for (int i = 0; i < 8; ++i) {
        float pc = __shfl_up(cs[i], 1, 32); if (lane == 0) pc = 0.f;
        rk[i] = al[i] / fmaxf(cs[i] - pc, EPS_);
        rs += rk[i]; cr[i] = rs;
    }
    vadd = scan_incl_add(rs, lane);
    eadd = __shfl_up(vadd, 1, 32); if (lane == 0) eadd = 0.f;
    #pragma unroll
    for (int i = 0; i < 8; ++i) cr[i] += eadd;
    // beta = eu * (cr[min(t+7,T-1)] - (cr[t]-rk[t]))
    float ncr[8];
    #pragma unroll
    for (int i = 0; i < 8; ++i) ncr[i] = __shfl_down(cr[i], 1, 32);
    float crlast = __shfl(cr[7], 31, 32);
    #pragma unroll
    for (int i = 0; i < 8; ++i) {
        float cidx = (i == 0) ? cr[7] : ((lane == 31) ? crlast : ncr[i - 1]);
        float bt = eu[i] * (cidx - (cr[i] - rk[i]));
        int t = t0 + i;
        beta[t * B_ + b]      = bt;
        alpha_new[t * B_ + b] = al[i];
    }
}

// ctx[b,e] = sum_t beta[t,b]*x[t,b,e] ; also pack z = [s ; ctx] as bf16.
__global__ void k_ctx_z(const float* __restrict__ beta, const float* __restrict__ x,
                        const float* __restrict__ s, float* __restrict__ ctx,
                        unsigned short* __restrict__ z_bf) {
    int b = blockIdx.x, e = threadIdx.x;               // 16 x 512
    float acc = 0.f;
    for (int t = 0; t < T_; ++t)
        acc += beta[t * B_ + b] * x[((size_t)(t * B_ + b)) * E_ + e];
    ctx[b * E_ + e] = acc;
    z_bf[b * 1024 + 512 + e] = f2bf(acc);
    z_bf[b * 1024 + e]       = f2bf(s[b * E_ + e]);
}

// ------------------------------- host side ---------------------------------

static inline dim3 gemm_grid(int N, int mtiles) {
    int nt    = (N + 15) / 16;      // 16-col tiles
    int pairs = (nt + 1) / 2;       // 2 tiles per wave
    return dim3((pairs + 3) / 4, mtiles);   // 4 waves per block
}

extern "C" void kernel_launch(void* const* d_in, const int* in_sizes, int n_in,
                              void* d_out, int out_size, void* d_ws, size_t ws_size,
                              hipStream_t stream) {
    const float* x         = (const float*)d_in[0];
    const float* att_mask  = (const float*)d_in[1];
    const float* emb_table = (const float*)d_in[2];
    const float* W_ih      = (const float*)d_in[3];
    const float* W_hh      = (const float*)d_in[4];
    const float* b_lstm    = (const float*)d_in[5];
    const float* W_proj    = (const float*)d_in[6];
    const float* Ws_m      = (const float*)d_in[7];
    const float* Wh_m      = (const float*)d_in[8];
    const float* v_m       = (const float*)d_in[9];
    const float* r_m       = (const float*)d_in[10];
    const float* Ws_c      = (const float*)d_in[11];
    const float* Wh_c      = (const float*)d_in[12];
    const float* v_c       = (const float*)d_in[13];
    const float* W_am      = (const float*)d_in[14];
    const float* W_lm      = (const float*)d_in[15];
    const int*   label     = (const int*)d_in[16];

    float* out_am = (float*)d_out;
    float* out_lm = out_am + (size_t)N_ * UB_;

    // bump allocator over workspace (256B aligned)
    char* base = (char*)d_ws; size_t off = 0;
    auto alloc = [&](size_t bytes) -> void* {
        off = (off + 255) & ~(size_t)255;
        void* r = base + off; off += bytes; return r;
    };
    unsigned short* x_bf     = (unsigned short*)alloc((size_t)T_*B_*E_ * 2);
    unsigned short* Wcat_bf  = (unsigned short*)alloc((size_t)4*H_*3072 * 2);
    unsigned short* Wproj_bf = (unsigned short*)alloc((size_t)E_*H_ * 2);
    unsigned short* Wsm_bf   = (unsigned short*)alloc((size_t)E_*E_ * 2);
    unsigned short* Whm_bf   = (unsigned short*)alloc((size_t)E_*E_ * 2);
    unsigned short* Wsc_bf   = (unsigned short*)alloc((size_t)E_*E_ * 2);
    unsigned short* Whc_bf   = (unsigned short*)alloc((size_t)E_*E_ * 2);
    unsigned short* Wam_bf   = (unsigned short*)alloc((size_t)N_*1024 * 2);
    unsigned short* Wlm_bf   = (unsigned short*)alloc((size_t)N_*E_ * 2);
    unsigned short* emb_bf   = (unsigned short*)alloc((size_t)UB_*E_ * 2);
    unsigned short* h_bf     = (unsigned short*)alloc((size_t)B_*H_ * 2);
    unsigned short* s_bf     = (unsigned short*)alloc((size_t)B_*E_ * 2);
    unsigned short* z_bf     = (unsigned short*)alloc((size_t)B_*1024 * 2);
    unsigned short* zin_bf   = (unsigned short*)alloc((size_t)B_*3072 * 2);
    float* xh_m   = (float*)alloc((size_t)T_*B_*E_ * 4);
    float* xh_c   = (float*)alloc((size_t)T_*B_*E_ * 4);
    float* gates  = (float*)alloc((size_t)B_*4*H_ * 4);
    float* h      = (float*)alloc((size_t)B_*H_ * 4);
    float* c      = (float*)alloc((size_t)B_*H_ * 4);
    float* ctx    = (float*)alloc((size_t)B_*E_ * 4);
    float* s      = (float*)alloc((size_t)B_*E_ * 4);
    float* sm     = (float*)alloc((size_t)B_*E_ * 4);
    float* sc     = (float*)alloc((size_t)B_*E_ * 4);
    float* e_raw  = (float*)alloc((size_t)T_*B_ * 4);
    float* u_raw  = (float*)alloc((size_t)T_*B_ * 4);
    float* alphaA = (float*)alloc((size_t)T_*B_ * 4);
    float* alphaB = (float*)alloc((size_t)T_*B_ * 4);
    float* beta   = (float*)alloc((size_t)T_*B_ * 4);
    (void)ws_size; (void)in_sizes; (void)n_in; (void)out_size;

    const dim3 blk128(128), blk256(256);

    // ---- one-time prep: bf16 weight conversions (loop-invariant) ----
    k_f32_to_bf16<<<2048, blk256, 0, stream>>>(x,      x_bf,     T_*B_*E_);
    k_f32_to_bf16<<<1024, blk256, 0, stream>>>(W_proj, Wproj_bf, E_*H_);
    k_f32_to_bf16<<<512,  blk256, 0, stream>>>(Ws_m,   Wsm_bf,   E_*E_);
    k_f32_to_bf16<<<512,  blk256, 0, stream>>>(Wh_m,   Whm_bf,   E_*E_);
    k_f32_to_bf16<<<512,  blk256, 0, stream>>>(Ws_c,   Wsc_bf,   E_*E_);
    k_f32_to_bf16<<<512,  blk256, 0, stream>>>(Wh_c,   Whc_bf,   E_*E_);
    k_f32_to_bf16<<<4096, blk256, 0, stream>>>(W_am,   Wam_bf,   N_*1024);
    k_f32_to_bf16<<<4096, blk256, 0, stream>>>(W_lm,   Wlm_bf,   N_*E_);
    k_build_wcat <<<4096, blk256, 0, stream>>>(W_ih, W_hh, Wcat_bf, 4*H_*3072);
    k_gather_emb <<<1536, blk256, 0, stream>>>(label, emb_table, emb_bf, UB_*E_);

    // ---- loop-invariant GEMMs: encoder projections + LM head ----
    k_gemm_bf16<<<gemm_grid(E_, T_*B_/16), blk128, 0, stream>>>(
        x_bf, E_, Whm_bf, E_, E_, E_, nullptr, xh_m, 0, E_, 0);
    k_gemm_bf16<<<gemm_grid(E_, T_*B_/16), blk128, 0, stream>>>(
        x_bf, E_, Whc_bf, E_, E_, E_, nullptr, xh_c, 0, E_, 0);
    k_gemm_bf16<<<gemm_grid(N_, U_), blk128, 0, stream>>>(
        emb_bf, E_, Wlm_bf, E_, N_, E_, nullptr, out_lm, 1, UB_, 0);

    k_init<<<128, blk256, 0, stream>>>(h, c, ctx, alphaA);

    float* aprev = alphaA;
    float* anew  = alphaB;
    for (int u = 0; u < U_; ++u) {
        k_build_zin<<<192, blk256, 0, stream>>>(emb_bf + (size_t)u * B_ * E_, ctx, h, zin_bf);
        // gates = zin @ [W_ih|W_hh]^T + b    (16 x 8192, K=3072)
        k_gemm_bf16<<<gemm_grid(4*H_, 1), blk128, 0, stream>>>(
            zin_bf, 3072, Wcat_bf, 3072, 4*H_, 3072, b_lstm, gates, 0, 4*H_, 0);
        k_lstm_pointwise<<<128, blk256, 0, stream>>>(gates, h, c, h_bf);
        // s = h @ W_proj^T    (16 x 512, K=2048)
        k_gemm_bf16<<<gemm_grid(E_, 1), blk128, 0, stream>>>(
            h_bf, H_, Wproj_bf, H_, E_, H_, nullptr, s, 0, E_, 0);
        k_f32_to_bf16<<<32, blk256, 0, stream>>>(s, s_bf, B_*E_);
        // sm = s @ Ws_m^T ; sc = s @ Ws_c^T
        k_gemm_bf16<<<gemm_grid(E_, 1), blk128, 0, stream>>>(
            s_bf, E_, Wsm_bf, E_, E_, E_, nullptr, sm, 0, E_, 0);
        k_gemm_bf16<<<gemm_grid(E_, 1), blk128, 0, stream>>>(
            s_bf, E_, Wsc_bf, E_, E_, E_, nullptr, sc, 0, E_, 0);
        k_energies<<<512, blk256, 0, stream>>>(sm, sc, xh_m, xh_c, v_m, v_c, e_raw, u_raw);
        k_alpha_beta<<<1, dim3(512), 0, stream>>>(e_raw, u_raw, att_mask, r_m, aprev, anew, beta);
        k_ctx_z<<<16, dim3(512), 0, stream>>>(beta, x, s, ctx, z_bf);
        // am_u = [s;ctx] @ W_am^T -> out[n, u*16+b]   (16 x 14839, K=1024)
        k_gemm_bf16<<<gemm_grid(N_, 1), blk128, 0, stream>>>(
            z_bf, 1024, Wam_bf, 1024, N_, 1024, nullptr, out_am, 1, UB_, u * B_);
        float* tmp = aprev; aprev = anew; anew = tmp;
    }
}
